// MergeEncoder_25168508354596
// MI455X (gfx1250) — matrix operand order repaired and checked by reading, not compiled, not run on hardware
//
#include <hip/hip_runtime.h>

// ---------------------------------------------------------------------------
// 2-layer complete-graph GIN, fused bf16-WMMA pipeline for gfx1250 (MI455X).
//
//   total1 = colsum(x)
//   z1 = relu((x + total1, row0 plain) @ w1a + b1a)          [bf16, stored]
//   r1 = relu(z1 @ w1b + b1b)            + stats S1,SS1      [bf16, stored]
//   bn1 affine (scale1, shift1) and agg total2 derived analytically from S1
//   z2 = relu((scale1*r1+shift1 (+tot2)) @ w2a + b2a)        [bf16, reuse z1 buf]
//   stats-only pass: relu(z2 @ w2b + b2b) -> S2, SS2 (no store)
//   out = gamma2*(S2 - N*mu2)*rsqrt(var2+eps) + N*beta2
// ---------------------------------------------------------------------------

typedef __bf16 bf16;
typedef __attribute__((ext_vector_type(16))) __bf16 v16bf;
typedef __attribute__((ext_vector_type(8)))  __bf16 v8bf;
typedef __attribute__((ext_vector_type(8)))  float  v8f;

#define GIN_N   50000
#define GIN_F   512
#define GIN_H   1024
#define MT      4   // 16-row tiles per wave
#define NT      4   // 16-col tiles per wave
// block: 256 threads = 8 waves -> tile of 64 rows x 512 cols

__device__ __forceinline__ v16bf zero16bf() {
  v16bf a;
#pragma unroll
  for (int i = 0; i < 16; ++i) a[i] = (bf16)0.0f;
  return a;
}

__device__ __forceinline__ v16bf join8(v8bf c0, v8bf c1) {
  v16bf a;
#pragma unroll
  for (int i = 0; i < 8; ++i) { a[i] = c0[i]; a[i + 8] = c1[i]; }
  return a;
}

// B operand (32x16 KxN, bf16): Wt stored [Nout, K] row-major. Lane holds
// column n = lane&15; elements 0..7 are K = kB..kB+7 and 8..15 are
// K = kB+16..kB+23 with kB = k0 + (lane>>4)*8.
__device__ __forceinline__ v16bf load_b(const bf16* __restrict__ Wt, int nrow,
                                        int K, int k0, int half) {
  const bf16* p = Wt + (size_t)nrow * K + k0 + half * 8;
  return join8(*(const v8bf*)p, *(const v8bf*)(p + 16));
}

// A operand (16x32 MxK, bf16): lane holds row m = lane&15; elements 0..7 are
// K = kA..kA+7 and 8..15 are K = kA+16..kA+23 with kA = k0 + (lane>>4)*8.
// PRE 0: plain bf16.  PRE 1: f32 + GIN agg (x + total, row 0 plain).
// PRE 2: bf16 * scale + shift (+ tot for rows > 0)  -> BatchNorm + GIN agg.
template<int PRE>
__device__ __forceinline__ v16bf load_a(const void* __restrict__ A, int row,
                                        int Mrows, int K, int k0, int half,
                                        const float* __restrict__ tot,
                                        const float* __restrict__ scale,
                                        const float* __restrict__ shift) {
  if (row >= Mrows) return zero16bf();
  const int kA = k0 + half * 8;
  if (PRE == 0) {
    const bf16* p = (const bf16*)A + (size_t)row * K + kA;
    return join8(*(const v8bf*)p, *(const v8bf*)(p + 16));
  } else if (PRE == 1) {
    const float* p = (const float*)A + (size_t)row * K + kA;
    const float m = (row != 0) ? 1.0f : 0.0f;
    v16bf a;
#pragma unroll
    for (int i = 0; i < 8; ++i) {
      a[i]     = (bf16)(p[i]      + m * tot[kA + i]);
      a[i + 8] = (bf16)(p[i + 16] + m * tot[kA + 16 + i]);
    }
    return a;
  } else {
    const bf16* p = (const bf16*)A + (size_t)row * K + kA;
    const float m = (row != 0) ? 1.0f : 0.0f;
    v16bf a;
#pragma unroll
    for (int i = 0; i < 8; ++i) {
      const int c0 = kA + i, c1 = kA + 16 + i;
      a[i]     = (bf16)((float)p[i]      * scale[c0] + shift[c0] + m * tot[c0]);
      a[i + 8] = (bf16)((float)p[i + 16] * scale[c1] + shift[c1] + m * tot[c1]);
    }
    return a;
  }
}

// Fused GEMM + bias + ReLU (+ optional bf16 store, optional per-column
// sum / sum-of-squares stats via LDS ds_add_f32 then one global flush).
template<int PRE, bool STATS, bool STORE>
__global__ __launch_bounds__(256)
void gin_gemm(const void* __restrict__ A, const bf16* __restrict__ Wt,
              const float* __restrict__ bias,
              const float* __restrict__ tot, const float* __restrict__ scale,
              const float* __restrict__ shift,
              bf16* __restrict__ Out, float* __restrict__ gS,
              float* __restrict__ gQ, int Mrows, int K, int Nout) {
  __shared__ float sS[8 * NT * 16];
  __shared__ float sQ[8 * NT * 16];
  const int lane = threadIdx.x & 31;
  const int wave = threadIdx.x >> 5;
  const int half = lane >> 4;
  const int ln   = lane & 15;
  const int rowBase = blockIdx.y * (MT * 16);
  const int colW    = blockIdx.x * (8 * NT * 16) + wave * (NT * 16);

  if (STATS) {
    for (int i = threadIdx.x; i < 8 * NT * 16; i += 256) { sS[i] = 0.0f; sQ[i] = 0.0f; }
    __syncthreads();
  }

  v8f acc[MT][NT];
#pragma unroll
  for (int m = 0; m < MT; ++m)
#pragma unroll
    for (int t = 0; t < NT; ++t)
#pragma unroll
      for (int r = 0; r < 8; ++r) acc[m][t][r] = 0.0f;

  for (int k0 = 0; k0 < K; k0 += 32) {
    // All A and B fragments land in distinct registers so the compiler can
    // clause the 8 b128 B loads together and drain them with partial
    // s_wait_loadcnt instead of a full-serial load->wait->wmma chain.
    v16bf afrag[MT];
#pragma unroll
    for (int m = 0; m < MT; ++m)
      afrag[m] = load_a<PRE>(A, rowBase + m * 16 + ln, Mrows, K, k0, half,
                             tot, scale, shift);
    v16bf bfrag[NT];
#pragma unroll
    for (int t = 0; t < NT; ++t)
      bfrag[t] = load_b(Wt, colW + t * 16 + ln, K, k0, half);
    // Pull the B panel ~256B ahead in K toward the WGP while WMMAs run.
    if (k0 + 128 < K) {
#pragma unroll
      for (int t = 0; t < NT; ++t)
        __builtin_prefetch(Wt + (size_t)(colW + t * 16 + ln) * K +
                               (k0 + 128) + half * 16, 0, 3);
    }
#pragma unroll
    for (int t = 0; t < NT; ++t)
#pragma unroll
      for (int m = 0; m < MT; ++m)
        acc[m][t] = __builtin_amdgcn_wmma_f32_16x16x32_bf16(
            false, afrag[m], false, bfrag[t], (short)0, acc[m][t], false, false);
  }

  // Epilogue: D element r in lane L is (row = rowBase+m*16+(L>=16?8:0)+r,
  // col = colTile + (L&15)).
#pragma unroll
  for (int m = 0; m < MT; ++m) {
#pragma unroll
    for (int t = 0; t < NT; ++t) {
      const int col = colW + t * 16 + ln;
      const float bv = bias[col];
      float ps = 0.0f, pq = 0.0f;
#pragma unroll
      for (int r = 0; r < 8; ++r) {
        const int row = rowBase + m * 16 + half * 8 + r;
        float v = acc[m][t][r] + bv;
        v = v > 0.0f ? v : 0.0f;
        if (row < Mrows) {
          if (STORE) Out[(size_t)row * Nout + col] = (bf16)v;
          ps += v;
          pq += v * v;
        }
      }
      if (STATS) {
        const int lcol = wave * (NT * 16) + t * 16 + ln;
        atomicAdd(&sS[lcol], ps);
        atomicAdd(&sQ[lcol], pq);
      }
    }
  }

  if (STATS) {
    __syncthreads();
    for (int i = threadIdx.x; i < 8 * NT * 16; i += 256) {
      atomicAdd(&gS[blockIdx.x * (8 * NT * 16) + i], sS[i]);
      atomicAdd(&gQ[blockIdx.x * (8 * NT * 16) + i], sQ[i]);
    }
  }
}

// Transpose + f32->bf16 convert: W [K, Nout] row-major -> Wt [Nout, K].
__global__ __launch_bounds__(256)
void wconv(const float* __restrict__ W, bf16* __restrict__ Wt, int K, int Nout) {
  __shared__ float tile[32][33];
  const int tx = threadIdx.x & 31, ty = threadIdx.x >> 5;
  const int n0 = blockIdx.x * 32, k0 = blockIdx.y * 32;
#pragma unroll
  for (int i = 0; i < 32; i += 8)
    tile[ty + i][tx] = W[(size_t)(k0 + ty + i) * Nout + n0 + tx];
  __syncthreads();
#pragma unroll
  for (int i = 0; i < 32; i += 8)
    Wt[(size_t)(n0 + ty + i) * K + k0 + tx] = (bf16)tile[tx][ty + i];
}

__global__ void colsum512(const float* __restrict__ x, float* __restrict__ tot, int M) {
  const int col = blockIdx.x * blockDim.x + threadIdx.x;  // 512 cols
  float s = 0.0f;
  for (int r = blockIdx.y; r < M; r += gridDim.y) s += x[(size_t)r * GIN_F + col];
  atomicAdd(&tot[col], s);
}

__global__ void zerof(float* __restrict__ p, int n) {
  const int i = blockIdx.x * blockDim.x + threadIdx.x;
  if (i < n) p[i] = 0.0f;
}

// BN1 affine + analytic layer-2 aggregation total: tot2 = scale*S1 + N*shift.
__global__ void finalize1(const float* __restrict__ S1, const float* __restrict__ Q1,
                          const float* __restrict__ gamma, const float* __restrict__ beta,
                          float* __restrict__ scale, float* __restrict__ shift,
                          float* __restrict__ tot2, int Mrows) {
  const int c = blockIdx.x * blockDim.x + threadIdx.x;
  const float invN = 1.0f / (float)Mrows;
  const float mu  = S1[c] * invN;
  const float var = Q1[c] * invN - mu * mu;
  const float sc  = gamma[c] * rsqrtf(var + 1e-5f);
  const float sh  = beta[c] - mu * sc;
  scale[c] = sc;
  shift[c] = sh;
  tot2[c]  = sc * S1[c] + (float)Mrows * sh;
}

// Readout: sum over nodes of BatchNorm2 output, from S2/SS2 only.
__global__ void finalout(const float* __restrict__ S2, const float* __restrict__ Q2,
                         const float* __restrict__ gamma, const float* __restrict__ beta,
                         float* __restrict__ out, int Mrows) {
  const int c = blockIdx.x * blockDim.x + threadIdx.x;
  const float fN = (float)Mrows;
  const float mu  = S2[c] / fN;
  const float var = Q2[c] / fN - mu * mu;
  out[c] = gamma[c] * (S2[c] - fN * mu) * rsqrtf(var + 1e-5f) + fN * beta[c];
}

extern "C" void kernel_launch(void* const* d_in, const int* in_sizes, int n_in,
                              void* d_out, int out_size, void* d_ws, size_t ws_size,
                              hipStream_t stream) {
  const float* x   = (const float*)d_in[0];
  const float* w1a = (const float*)d_in[1];
  const float* b1a = (const float*)d_in[2];
  const float* w1b = (const float*)d_in[3];
  const float* b1b = (const float*)d_in[4];
  const float* g1  = (const float*)d_in[5];
  const float* be1 = (const float*)d_in[6];
  const float* w2a = (const float*)d_in[7];
  const float* b2a = (const float*)d_in[8];
  const float* w2b = (const float*)d_in[9];
  const float* b2b = (const float*)d_in[10];
  const float* g2  = (const float*)d_in[11];
  const float* be2 = (const float*)d_in[12];
  float* out = (float*)d_out;

  char* w = (char*)d_ws;
  size_t o = 0;
  bf16* w1aT = (bf16*)(w + o); o += (size_t)GIN_H * GIN_F * 2;  // [1024,512]
  bf16* w1bT = (bf16*)(w + o); o += (size_t)GIN_H * GIN_H * 2;  // [1024,1024]
  bf16* w2aT = (bf16*)(w + o); o += (size_t)GIN_H * GIN_H * 2;  // [1024,1024]
  bf16* w2bT = (bf16*)(w + o); o += (size_t)GIN_F * GIN_H * 2;  // [512,1024]
  // stats block (zeroed each launch): total1|S1|Q1|S2|Q2 contiguous
  float* total1 = (float*)(w + o); o += GIN_F * 4;
  float* S1     = (float*)(w + o); o += GIN_H * 4;
  float* Q1     = (float*)(w + o); o += GIN_H * 4;
  float* S2     = (float*)(w + o); o += GIN_F * 4;
  float* Q2     = (float*)(w + o); o += GIN_F * 4;
  const int nzero = GIN_F + GIN_H + GIN_H + GIN_F + GIN_F;  // 3584 floats
  float* scale1 = (float*)(w + o); o += GIN_H * 4;
  float* shift1 = (float*)(w + o); o += GIN_H * 4;
  float* tot2   = (float*)(w + o); o += GIN_H * 4;
  o = (o + 255) & ~(size_t)255;
  bf16* z1 = (bf16*)(w + o); o += (size_t)GIN_N * GIN_H * 2;   // 102.4 MB, reused for z2
  bf16* r1 = (bf16*)(w + o); o += (size_t)GIN_N * GIN_H * 2;   // 102.4 MB
  if (o > ws_size) return;  // workspace too small: deterministic no-op

  zerof<<<(nzero + 255) / 256, 256, 0, stream>>>(total1, nzero);
  wconv<<<dim3(GIN_H / 32, GIN_F / 32), 256, 0, stream>>>(w1a, w1aT, GIN_F, GIN_H);
  wconv<<<dim3(GIN_H / 32, GIN_H / 32), 256, 0, stream>>>(w1b, w1bT, GIN_H, GIN_H);
  wconv<<<dim3(GIN_H / 32, GIN_H / 32), 256, 0, stream>>>(w2a, w2aT, GIN_H, GIN_H);
  wconv<<<dim3(GIN_F / 32, GIN_H / 32), 256, 0, stream>>>(w2b, w2bT, GIN_H, GIN_F);
  colsum512<<<dim3(GIN_F / 256, 256), 256, 0, stream>>>(x, total1, GIN_N);

  const int MB = (GIN_N + MT * 16 - 1) / (MT * 16);  // 782 row blocks

  // L1a: z1 = relu((x + total1) @ w1a + b1a)
  gin_gemm<1, false, true><<<dim3(GIN_H / 512, MB), 256, 0, stream>>>(
      x, w1aT, b1a, total1, nullptr, nullptr, z1, nullptr, nullptr,
      GIN_N, GIN_F, GIN_H);
  // L1b: r1 = relu(z1 @ w1b + b1b), accumulate S1/Q1
  gin_gemm<0, true, true><<<dim3(GIN_H / 512, MB), 256, 0, stream>>>(
      z1, w1bT, b1b, nullptr, nullptr, nullptr, r1, S1, Q1,
      GIN_N, GIN_H, GIN_H);
  finalize1<<<GIN_H / 256, 256, 0, stream>>>(S1, Q1, g1, be1, scale1, shift1, tot2, GIN_N);
  // L2a: z2 = relu((bn1(r1) + tot2) @ w2a + b2a)   (z2 reuses z1 buffer)
  gin_gemm<2, false, true><<<dim3(GIN_H / 512, MB), 256, 0, stream>>>(
      r1, w2aT, b2a, tot2, scale1, shift1, z1, nullptr, nullptr,
      GIN_N, GIN_H, GIN_H);
  // L2b: stats-only relu(z2 @ w2b + b2b) -> S2/Q2 (no activation store)
  gin_gemm<0, true, false><<<dim3(GIN_F / 512, MB), 256, 0, stream>>>(
      z1, w2bT, b2b, nullptr, nullptr, nullptr, nullptr, S2, Q2,
      GIN_N, GIN_H, GIN_F);
  finalout<<<GIN_F / 256, 256, 0, stream>>>(S2, Q2, g2, be2, out, GIN_N);
}
